// NodeBlock_1477468750139
// MI455X (gfx1250) — matrix operand order repaired and checked by reading, not compiled, not run on hardware
//
#include <hip/hip_runtime.h>
#include <hip/hip_bf16.h>
#include <stdint.h>

#define NODES   50000
#define DFEAT   128
#define EEDGES  600000

typedef __attribute__((ext_vector_type(16))) __bf16        v16bf;
typedef __attribute__((ext_vector_type(8)))  float         v8f;
typedef __attribute__((ext_vector_type(8)))  unsigned int  v8u;
typedef __attribute__((ext_vector_type(4)))  float         f4v;   // native vector for NT loads

// round-to-nearest-even fp32 -> bf16 bits
__device__ __forceinline__ unsigned short f2bf(float f) {
    unsigned int u = __builtin_bit_cast(unsigned int, f);
    unsigned int r = u + 0x7FFFu + ((u >> 16) & 1u);
    return (unsigned short)(r >> 16);
}
__device__ __forceinline__ float bf2f(unsigned short s) {
    unsigned int u = ((unsigned int)s) << 16;
    return __builtin_bit_cast(float, u);
}
__device__ __forceinline__ unsigned int pack2bf(float lo_elem, float hi_elem) {
    return ((unsigned int)f2bf(hi_elem) << 16) | (unsigned int)f2bf(lo_elem);
}

// Build a v16bf fragment from 8 uints located at p[0..3] and p[8..11].
__device__ __forceinline__ v16bf frag_gap(const unsigned int* p) {
    const uint4* q = (const uint4*)p;
    uint4 g0 = q[0];
    uint4 g1 = q[2];           // +8 uints = +16 K values
    v8u u;
    u[0] = g0.x; u[1] = g0.y; u[2] = g0.z; u[3] = g0.w;
    u[4] = g1.x; u[5] = g1.y; u[6] = g1.z; u[7] = g1.w;
    return __builtin_bit_cast(v16bf, u);
}
// Build a v16bf fragment from 8 consecutive uints.
__device__ __forceinline__ v16bf frag_seq(const unsigned int* p) {
    const uint4* q = (const uint4*)p;
    uint4 g0 = q[0];
    uint4 g1 = q[1];
    v8u u;
    u[0] = g0.x; u[1] = g0.y; u[2] = g0.z; u[3] = g0.w;
    u[4] = g1.x; u[5] = g1.y; u[6] = g1.z; u[7] = g1.w;
    return __builtin_bit_cast(v16bf, u);
}

// ---------------- kernel 0: pack weights as split-bf16, column-major K-pairs ---
// w1h/w1l: [n][kp] kp=0..127 packing K=2kp,2kp+1 of W1 (K=256 rows, 128 cols)
// w2h/w2l: [n][kp] kp=0..63                  of W2 (K=128 rows, 128 cols)
__global__ void pack_weights_kernel(const float* __restrict__ W1,
                                    const float* __restrict__ W2,
                                    unsigned int* __restrict__ w1h,
                                    unsigned int* __restrict__ w1l,
                                    unsigned int* __restrict__ w2h,
                                    unsigned int* __restrict__ w2l) {
    int i = blockIdx.x * blockDim.x + threadIdx.x;
    if (i < 128 * 128) {
        int n = i >> 7, kp = i & 127;
        float a = W1[(size_t)(2 * kp) * DFEAT + n];
        float b = W1[(size_t)(2 * kp + 1) * DFEAT + n];
        unsigned short ah = f2bf(a), bh = f2bf(b);
        w1h[n * 128 + kp] = ((unsigned int)bh << 16) | ah;
        w1l[n * 128 + kp] = pack2bf(a - bf2f(ah), b - bf2f(bh));
    } else if (i < 128 * 128 + 128 * 64) {
        int j = i - 128 * 128;
        int n = j >> 6, kp = j & 63;
        float a = W2[(size_t)(2 * kp) * DFEAT + n];
        float b = W2[(size_t)(2 * kp + 1) * DFEAT + n];
        unsigned short ah = f2bf(a), bh = f2bf(b);
        w2h[n * 64 + kp] = ((unsigned int)bh << 16) | ah;
        w2l[n * 64 + kp] = pack2bf(a - bf2f(ah), b - bf2f(bh));
    }
}

// ---------------- kernel 1: zero the aggregation buffer ----------------
// regular-temporal stores: agg lines should allocate and stay in L2 for atomics
__global__ void zero_agg_kernel(float4* __restrict__ agg4, int n4) {
    int i = blockIdx.x * blockDim.x + threadIdx.x;
    if (i < n4) agg4[i] = make_float4(0.f, 0.f, 0.f, 0.f);
}

// ---------------- kernel 2: scatter-add edge features ----------------
// one wave = one edge row (lanes cover 128 floats as 32 float4 quads).
// edge_attr is 307MB streamed exactly once -> non-temporal loads, so the
// 192MB L2 keeps the 25.6MB agg buffer (atomic destinations) resident.
__global__ void scatter_add_kernel(const f4v* __restrict__ ea4,
                                   const int* __restrict__ recv,
                                   float* __restrict__ agg) {
    int i = blockIdx.x * blockDim.x + threadIdx.x;   // over EEDGES * 32 quads
    if (i >= EEDGES * (DFEAT / 4)) return;
    int edge = i >> 5;                 // wave-uniform
    int q    = i & 31;
    f4v v = __builtin_nontemporal_load(&ea4[(size_t)edge * 32 + q]);
    float* dst = agg + (size_t)recv[edge] * DFEAT + q * 4;
    unsafeAtomicAdd(dst + 0, v.x);
    unsafeAtomicAdd(dst + 1, v.y);
    unsafeAtomicAdd(dst + 2, v.z);
    unsafeAtomicAdd(dst + 3, v.w);
}

// ---------------- kernel 3: fused MLP via split-bf16 WMMA ----------------
// 1 block = 256 threads = 8 waves = 16 node rows; wave w owns cols [16w,16w+16).
// Split precision: a*b ~ ahi*bhi + ahi*blo + alo*bhi, fp32 accumulate.
__global__ __launch_bounds__(256)
void mlp_fused_kernel(const float* __restrict__ x,
                      const float* __restrict__ agg,
                      const unsigned int* __restrict__ w1h,
                      const unsigned int* __restrict__ w1l,
                      const unsigned int* __restrict__ w2h,
                      const unsigned int* __restrict__ w2l,
                      const float* __restrict__ b1,
                      const float* __restrict__ b2,
                      float* __restrict__ out) {
    __shared__ unsigned short sCollH[16 * 256];   // [x||agg] hi bf16 (K contiguous)
    __shared__ unsigned short sCollL[16 * 256];   // residual lo bf16
    __shared__ unsigned short sHH[16 * 128];      // hidden hi
    __shared__ unsigned short sHL[16 * 128];      // hidden lo

    const int t     = threadIdx.x;
    const int lane  = t & 31;
    const int wave  = t >> 5;            // N-tile index 0..7
    const int node0 = blockIdx.x * 16;   // NODES = 3125 * 16 exactly

    // stage collected = [x || agg] rows as split bf16 (float4 loads, uint2 stores)
    #pragma unroll
    for (int it = 0; it < 4; ++it) {
        int idx4 = t + it * 256;               // 1024 float4 chunks
        int row  = idx4 >> 6;
        int c4   = (idx4 & 63) * 4;
        size_t node = (size_t)(node0 + row);
        float4 v = (c4 < DFEAT)
                     ? *(const float4*)&x[node * DFEAT + c4]
                     : *(const float4*)&agg[node * DFEAT + (c4 - DFEAT)];
        unsigned short h0 = f2bf(v.x), h1 = f2bf(v.y), h2 = f2bf(v.z), h3 = f2bf(v.w);
        uint2 hv, lv;
        hv.x = ((unsigned int)h1 << 16) | h0;
        hv.y = ((unsigned int)h3 << 16) | h2;
        lv.x = pack2bf(v.x - bf2f(h0), v.y - bf2f(h1));
        lv.y = pack2bf(v.z - bf2f(h2), v.w - bf2f(h3));
        *(uint2*)&sCollH[row * 256 + c4] = hv;
        *(uint2*)&sCollL[row * 256 + c4] = lv;
    }
    __syncthreads();

    const int half = lane >> 4;                 // 0: lanes 0-15, 1: lanes 16-31
    const int arow = lane & 15;                 // A-matrix row (M)
    const int bcol = (lane & 15) + wave * 16;   // output column

    v8f acc = {0.f, 0.f, 0.f, 0.f, 0.f, 0.f, 0.f, 0.f};

    // ---- GEMM1: K = 256 (8 K-tiles of 32) ----
    #pragma unroll
    for (int kt = 0; kt < 8; ++kt) {
        const unsigned int* aph = (const unsigned int*)&sCollH[arow * 256 + kt * 32 + half * 8];
        const unsigned int* apl = (const unsigned int*)&sCollL[arow * 256 + kt * 32 + half * 8];
        v16bf aH = frag_gap(aph);
        v16bf aL = frag_gap(apl);
        const unsigned int* bph = &w1h[bcol * 128 + kt * 16 + half * 4];
        const unsigned int* bpl = &w1l[bcol * 128 + kt * 16 + half * 4];
        v16bf bH = frag_gap(bph);
        v16bf bL = frag_gap(bpl);
        acc = __builtin_amdgcn_wmma_f32_16x16x32_bf16(false, aL, false, bH, (short)0, acc, false, false);
        acc = __builtin_amdgcn_wmma_f32_16x16x32_bf16(false, aH, false, bL, (short)0, acc, false, false);
        acc = __builtin_amdgcn_wmma_f32_16x16x32_bf16(false, aH, false, bH, (short)0, acc, false, false);
    }

    // bias + ReLU, stage hidden (split bf16) to LDS
    {
        float bias1 = b1[bcol];
        #pragma unroll
        for (int r = 0; r < 8; ++r) {
            int m = r + half * 8;                // C/D layout: lanes>=16 -> M+8
            float h = fmaxf(acc[r] + bias1, 0.f);
            unsigned short hh = f2bf(h);
            sHH[m * 128 + bcol] = hh;
            sHL[m * 128 + bcol] = f2bf(h - bf2f(hh));
        }
    }
    __syncthreads();

    v8f acc2 = {0.f, 0.f, 0.f, 0.f, 0.f, 0.f, 0.f, 0.f};

    // ---- GEMM2: K = 128 (4 K-tiles of 32) ----
    #pragma unroll
    for (int kt = 0; kt < 4; ++kt) {
        const unsigned int* aph = (const unsigned int*)&sHH[arow * 128 + kt * 32 + half * 8];
        const unsigned int* apl = (const unsigned int*)&sHL[arow * 128 + kt * 32 + half * 8];
        v16bf aH = frag_gap(aph);
        v16bf aL = frag_gap(apl);
        const unsigned int* bph = &w2h[bcol * 64 + kt * 16 + half * 4];
        const unsigned int* bpl = &w2l[bcol * 64 + kt * 16 + half * 4];
        v16bf bH = frag_gap(bph);
        v16bf bL = frag_gap(bpl);
        acc2 = __builtin_amdgcn_wmma_f32_16x16x32_bf16(false, aL, false, bH, (short)0, acc2, false, false);
        acc2 = __builtin_amdgcn_wmma_f32_16x16x32_bf16(false, aH, false, bL, (short)0, acc2, false, false);
        acc2 = __builtin_amdgcn_wmma_f32_16x16x32_bf16(false, aH, false, bH, (short)0, acc2, false, false);
    }

    // out is written once and never re-read on device -> non-temporal stores
    {
        float bias2 = b2[bcol];
        #pragma unroll
        for (int r = 0; r < 8; ++r) {
            int m = r + half * 8;
            __builtin_nontemporal_store(acc2[r] + bias2,
                                        &out[(size_t)(node0 + m) * DFEAT + bcol]);
        }
    }
}

extern "C" void kernel_launch(void* const* d_in, const int* in_sizes, int n_in,
                              void* d_out, int out_size, void* d_ws, size_t ws_size,
                              hipStream_t stream) {
    (void)in_sizes; (void)n_in; (void)out_size; (void)ws_size;
    const float* x         = (const float*)d_in[0];
    const float* edge_attr = (const float*)d_in[1];
    const int*   recv      = (const int*)  d_in[2];
    const float* W1        = (const float*)d_in[3];
    const float* b1        = (const float*)d_in[4];
    const float* W2        = (const float*)d_in[5];
    const float* b2        = (const float*)d_in[6];
    float* out = (float*)d_out;

    // workspace layout: agg (25.6MB) | w1h | w1l | w2h | w2l (packed split-bf16)
    float* agg = (float*)d_ws;
    unsigned int* w1h = (unsigned int*)((char*)d_ws + (size_t)NODES * DFEAT * sizeof(float));
    unsigned int* w1l = w1h + 128 * 128;
    unsigned int* w2h = w1l + 128 * 128;
    unsigned int* w2l = w2h + 128 * 64;

    // 0) pack weights (once per launch; removes per-block conversion VALU)
    int npack = 128 * 128 + 128 * 64;
    pack_weights_kernel<<<(npack + 255) / 256, 256, 0, stream>>>(W1, W2, w1h, w1l, w2h, w2l);

    // 1) zero agg
    int n4 = NODES * DFEAT / 4;
    zero_agg_kernel<<<(n4 + 255) / 256, 256, 0, stream>>>((float4*)agg, n4);

    // 2) scatter-add edges (NT streaming reads, fp32 atomics into L2-resident agg)
    int total = EEDGES * (DFEAT / 4);
    scatter_add_kernel<<<(total + 255) / 256, 256, 0, stream>>>(
        (const f4v*)edge_attr, recv, agg);

    // 3) fused MLP with split-bf16 WMMA
    mlp_fused_kernel<<<NODES / 16, 256, 0, stream>>>(
        x, agg, w1h, w1l, w2h, w2l, b1, b2, out);
}
